// MPCPlanner_43765716746764
// MI455X (gfx1250) — matrix operand order repaired and checked by reading, compiled, not run on hardware
//
#include <hip/hip_runtime.h>
#include <stdint.h>

// ---------------------------------------------------------------------------
// CEM MPC planner for MI455X (gfx1250, wave32, WMMA).
// Both rollout steps are pure bf16 WMMA GEMMs with f32 accumulation:
//   t=0: z1 = tanh([a0|1] @ [Wa ; zWz0_b])            (K=32, 1 k-tile)
//   t=1: z2 = tanh([z1|a1|1|0] @ [Wz ; Wa ; bias ; 0]) (K=544, 17 k-tiles)
// Action/bias terms are folded into the GEMM K dimension; the epilogue is
// branch-free fast tanh (v_exp/v_rcp) + |z2-goal| accumulation, so EXEC stays
// all-ones through the WMMA stream. W' (544KB bf16) is L2-resident.
// ---------------------------------------------------------------------------

#define NB 8        // batch
#define ND 512      // latent dim
#define NS 4096     // samples
#define NH 2        // horizon
#define NA 7        // action dim
#define NITERS 10
#define TOPK 10

#define KAUG 544               // 512 + 7 (a1) + 1 (bias sel) + 24 pad -> 17 k-tiles
#define KT1  17                // k-tiles for t=1 GEMM
#define S_PER_BLK 32           // samples per rollout block
#define ROLL_THREADS 64        // 2 waves; wave w owns rows [16w, 16w+16)
#define PITCH 552              // bf16 elems per LDS row (1104B: 16B-aligned, banks spread)

typedef __attribute__((ext_vector_type(16))) __bf16 v16bf;
typedef __attribute__((ext_vector_type(8)))  float  v8f;
typedef __attribute__((ext_vector_type(4)))  unsigned int v4u;

union Frag16 { v16bf v; v4u q[2]; unsigned short s[16]; };

__device__ __forceinline__ unsigned short f32_to_bf16(float f) {
  unsigned int u = __float_as_uint(f);
  u += 0x7FFFu + ((u >> 16) & 1u);        // round-to-nearest-even
  return (unsigned short)(u >> 16);
}

// Branch-free tanh: sign(x) * (1 - 2/(2^(2*log2e*|x|)+1)).
// v_exp_f32 + v_rcp_f32; saturates correctly for large |x| (exp->inf, rcp->0).
__device__ __forceinline__ float fast_tanh(float x) {
  float ax = __builtin_fabsf(x);
  float e  = __builtin_amdgcn_exp2f(ax * 2.88539008177792681472f); // 2*log2(e)
  float t  = 1.0f - 2.0f * __builtin_amdgcn_rcpf(e + 1.0f);
  return __builtin_copysignf(t, x);
}

__device__ __forceinline__ uint64_t splitmix64(uint64_t x) {
  x += 0x9E3779B97F4A7C15ULL;
  x = (x ^ (x >> 30)) * 0xBF58476D1CE4E5B9ULL;
  x = (x ^ (x >> 27)) * 0x94D049BB133111EBULL;
  return x ^ (x >> 31);
}

// Deterministic counter-based standard normal (Box-Muller), branch-free:
// sqrt(-2*ln2*log2(u1)) * cos(2*pi*u2); v_cos_f32 takes revolutions directly.
__device__ __forceinline__ float gauss_from_counter(uint64_t ctr) {
  uint64_t h1 = splitmix64(ctr * 2ULL + 0x0123456789ABCDEFULL);
  uint64_t h2 = splitmix64(ctr * 2ULL + 1ULL + 0x0123456789ABCDEFULL);
  float u1 = ((float)((unsigned)(h1 >> 40) + 1u)) * (1.0f / 16777217.0f); // (0,1]
  float u2 = ((float)((unsigned)(h2 >> 40))) * (1.0f / 16777216.0f);     // [0,1)
  float r = __builtin_amdgcn_sqrtf(-1.38629436111989062f *               // 2*ln2
                                   __builtin_amdgcn_logf(u1));
  return r * __builtin_amdgcn_cosf(u2);
}

// ---------------------------------------------------------------------------
// Pack augmented W' = [Wz(512) ; Wa(7) ; bias(1) ; 0(24)] (K=544, N=512) into
// bf16 B-fragments in contiguous k order (A uses the same): fragment (kt,nt):
// lane L<16 holds e[k]=B[kt*32+k][nt*16+L], lane L>=16 holds e[k]=B[kt*32+16+k][..].
// Each lane's 16 bf16 stored contiguously (32B) -> two b128 loads.
// ---------------------------------------------------------------------------
__global__ __launch_bounds__(256) void prep_wz_kernel(const float* __restrict__ Wz,
                                                      const float* __restrict__ Wa,
                                                      const float* __restrict__ bias,
                                                      unsigned short* __restrict__ WzP) {
  int tid = blockIdx.x * 256 + threadIdx.x;      // 17*32*32*16 = 278528 total
  int k    = tid & 15;
  int lane = (tid >> 4) & 31;
  int nt   = (tid >> 9) & 31;
  int kt   = tid >> 14;                          // 0..16
  int row = kt * 32 + ((lane & 16) ? 16 + k : k);
  int col = nt * 16 + (lane & 15);
  float v;
  if (row < ND)            v = Wz[row * ND + col];
  else if (row < ND + NA)  v = Wa[(row - ND) * ND + col];
  else if (row == ND + NA) v = bias[col];
  else                     v = 0.0f;
  WzP[tid] = f32_to_bf16(v);
}

// zWz0[b][d] = z_current[b] . Wz[:,d] + bias[d]; also reset mean/std state.
__global__ __launch_bounds__(512) void init_kernel(const float* __restrict__ z_current,
                                                   const float* __restrict__ Wz,
                                                   const float* __restrict__ bias,
                                                   float* __restrict__ zWz0,
                                                   float* __restrict__ actMean,
                                                   float* __restrict__ actStd) {
  int b = blockIdx.x, d = threadIdx.x;
  float acc = bias[d];
  for (int kk = 0; kk < ND; ++kk) acc += z_current[b * ND + kk] * Wz[kk * ND + d];
  zWz0[b * ND + d] = acc;
  if (d < NH * NA) { actMean[b * NH * NA + d] = 0.0f; actStd[b * NH * NA + d] = 0.05f; }
}

// Pack per-batch t=0 B matrix B0_b = [Wa(7) ; zWz0_b(1) ; 0(24)] (K=32, N=512)
// into bf16 fragments (1 k-tile x 32 n-tiles per batch).
__global__ __launch_bounds__(256) void pack_b0_kernel(const float* __restrict__ Wa,
                                                      const float* __restrict__ zWz0,
                                                      unsigned short* __restrict__ B0P) {
  int tid = blockIdx.x * 256 + threadIdx.x;      // 8*32*32*16 = 131072 total
  int k    = tid & 15;
  int lane = (tid >> 4) & 31;
  int nt   = (tid >> 9) & 31;
  int b    = tid >> 14;
  int lr   = (lane & 16) ? 16 + k : k;           // row within K=32
  int col  = nt * 16 + (lane & 15);
  float v;
  if (lr < NA)       v = Wa[lr * ND + col];
  else if (lr == NA) v = zWz0[b * ND + col];
  else               v = 0.0f;
  B0P[tid] = f32_to_bf16(v);
}

// ---------------------------------------------------------------------------
// Fused sample + rollout + energy kernel. One block = 32 samples of batch b.
// Wave w owns rows [16w,16w+16): 32 WMMA for t=0, 544 WMMA for t=1.
// ---------------------------------------------------------------------------
__global__ __launch_bounds__(ROLL_THREADS) void rollout_kernel(
    const float* __restrict__ z_goal,      // [B][D]
    const unsigned short* __restrict__ WzP,// packed bf16 W' fragments
    const unsigned short* __restrict__ B0P,// packed bf16 t=0 B fragments
    const float* __restrict__ actMean,     // [B][H][A]
    const float* __restrict__ actStd,      // [B][H][A]
    float* __restrict__ samples,           // [B][S][H][A]
    float* __restrict__ energies,          // [B][S]
    int it) {
  __shared__ unsigned short z1[S_PER_BLK * PITCH];   // 35,328 B
  __shared__ float actL[S_PER_BLK][NH][8];           //  2,048 B (pad 7->8)
  __shared__ float goalL[ND];                        //  2,048 B

  const int b    = blockIdx.x >> 7;                  // 128 blocks / batch
  const int s0   = (blockIdx.x & 127) * S_PER_BLK;
  const int tid  = threadIdx.x;
  const int lane = tid & 31;
  const int wave = tid >> 5;

  for (int i = tid; i < ND; i += ROLL_THREADS) goalL[i] = z_goal[b * ND + i];

  // ---- sample generation + clipping (deterministic per (it,b,s,t,a)) ----
  for (int i = tid; i < S_PER_BLK * NH * NA; i += ROLL_THREADS) {
    int s = i / (NH * NA);
    int ta = i % (NH * NA);
    int t = ta / NA, a = ta % NA;
    uint64_t ctr = (((uint64_t)(it * NB + b) * NS + (uint64_t)(s0 + s)) * (NH * NA)) + (uint64_t)ta;
    float eps = gauss_from_counter(ctr);
    int mi = (b * NH + t) * NA + a;
    float v = actMean[mi] + actStd[mi] * eps;
    if (a < 3)      v = fminf(fmaxf(v, -0.05f), 0.05f);
    else if (a < 6) v = 0.0f;
    else            v = fminf(fmaxf(v, -0.75f), 0.75f);
    actL[s][t][a] = v;
    samples[(((size_t)b * NS + (size_t)(s0 + s)) * NH + t) * NA + a] = v;
  }
  __syncthreads();

  const int rbase   = wave * 16;
  const int halfsel = (lane >> 4) & 1;      // 0: lanes 0-15, 1: lanes 16-31
  const int lcol    = lane & 15;

  // ---- t=0: z1 = tanh([a0|1] @ [Wa ; zWz0_b]) via WMMA, bf16 into LDS ----
  // A fragment built in registers: lanes<16 carry K=0..7 = {a0[0..6], 1};
  // their K=8..15 and all of lanes>=16 (K=16..31) are zero.
  Frag16 a0f;
#pragma unroll
  for (int e = 0; e < 16; ++e) a0f.s[e] = 0;
  if (!halfsel) {
#pragma unroll
    for (int a = 0; a < NA; ++a) a0f.s[a] = f32_to_bf16(actL[rbase + lcol][0][a]);
    a0f.s[NA] = 0x3F80;                     // 1.0 in bf16
  }
  for (int nt = 0; nt < 32; ++nt) {
    const unsigned short* bp = B0P + (size_t)(((b * 32 + nt) * 32 + lane) * 16);
    Frag16 bf;
    bf.q[0] = *(const v4u*)(bp);
    bf.q[1] = *(const v4u*)(bp + 8);
    v8f acc = (v8f){0.f,0.f,0.f,0.f,0.f,0.f,0.f,0.f};
    acc = __builtin_amdgcn_wmma_f32_16x16x32_bf16(false, a0f.v, false, bf.v,
                                                  (short)0, acc, false, false);
#pragma unroll
    for (int v = 0; v < 8; ++v) {
      int row = rbase + v + 8 * halfsel;    // C layout: slot v -> row v/v+8, col=lcol
      z1[row * PITCH + nt * 16 + lcol] = f32_to_bf16(fast_tanh(acc[v]));
    }
  }

  // ---- fill augmented K region of own wave's rows: [a1(7) | 1 | 0(24)] ----
  for (int i = lane; i < 16 * 32; i += 32) {
    int r = rbase + (i >> 5);
    int c = ND + (i & 31);
    float v = (c < ND + NA) ? actL[r][1][c - ND] : ((c == ND + NA) ? 1.0f : 0.0f);
    z1[r * PITCH + c] = f32_to_bf16(v);
  }
  __syncthreads();

  // ---- t=1: z2 = tanh(z1' @ W'); energy = mean|z2 - goal| ----
  float esum[8];
#pragma unroll
  for (int v = 0; v < 8; ++v) esum[v] = 0.0f;

  for (int nb = 0; nb < 8; ++nb) {          // 64-column output chunks
    v8f acc[4];
#pragma unroll
    for (int j = 0; j < 4; ++j) acc[j] = (v8f){0.f,0.f,0.f,0.f,0.f,0.f,0.f,0.f};

    for (int kt = 0; kt < KT1; ++kt) {      // K in steps of 32 (17 tiles)
      // A fragment (16x32 bf16): lane<16 -> K {0..15}; lane>=16 -> {16..31}
      Frag16 af;
      const unsigned short* zr = &z1[(rbase + lcol) * PITCH + kt * 32 + (halfsel ? 16 : 0)];
      af.q[0] = *(const v4u*)(zr);
      af.q[1] = *(const v4u*)(zr + 8);
#pragma unroll
      for (int j = 0; j < 4; ++j) {
        int nt = nb * 4 + j;
        const unsigned short* bp = WzP + (size_t)(((kt * 32 + nt) * 32 + lane) * 16);
        if (kt + 1 < KT1 && j == 0)
          __builtin_prefetch(WzP + (size_t)((((kt + 1) * 32 + nt) * 32 + lane) * 16), 0, 1);
        Frag16 bf;
        bf.q[0] = *(const v4u*)(bp);
        bf.q[1] = *(const v4u*)(bp + 8);
        acc[j] = __builtin_amdgcn_wmma_f32_16x16x32_bf16(
            /*neg_a=*/false, af.v, /*neg_b=*/false, bf.v,
            /*c_mod=*/(short)0, acc[j], /*reuse_a=*/false, /*reuse_b=*/false);
      }
    }

    // epilogue: branch-free tanh + L1 accumulation (actions/bias in GEMM)
#pragma unroll
    for (int j = 0; j < 4; ++j) {
      int col = (nb * 4 + j) * 16 + lcol;
      float gl = goalL[col];
#pragma unroll
      for (int v = 0; v < 8; ++v) esum[v] += __builtin_fabsf(fast_tanh(acc[j][v]) - gl);
    }
  }

  // cross-lane reduction within each 16-lane half (rows differ per half)
#pragma unroll
  for (int v = 0; v < 8; ++v) {
    float e = esum[v];
    e += __shfl_xor(e, 1, 16);
    e += __shfl_xor(e, 2, 16);
    e += __shfl_xor(e, 4, 16);
    e += __shfl_xor(e, 8, 16);
    if (lcol == 0) {
      int row = rbase + v + 8 * halfsel;
      energies[(size_t)b * NS + (size_t)(s0 + row)] = e * (1.0f / (float)ND);
    }
  }
}

// ---------------------------------------------------------------------------
// Per-batch top-k (10 argmin passes over LDS-cached energies) + momentum update.
// ---------------------------------------------------------------------------
__global__ __launch_bounds__(256) void topk_update_kernel(const float* __restrict__ energies,
                                                          const float* __restrict__ samples,
                                                          float* __restrict__ actMean,
                                                          float* __restrict__ actStd) {
  int b = blockIdx.x, tid = threadIdx.x;
  __shared__ float eL[NS];
  __shared__ float rv[256];
  __shared__ int   ri[256];
  __shared__ int   selI[TOPK];

  for (int i = tid; i < NS; i += 256) eL[i] = energies[(size_t)b * NS + i];
  __syncthreads();

  for (int k = 0; k < TOPK; ++k) {
    float best = 3.0e38f; int bi = 0;
    for (int i = tid; i < NS; i += 256) {
      float x = eL[i];
      if (x < best) { best = x; bi = i; }
    }
    rv[tid] = best; ri[tid] = bi;
    __syncthreads();
    for (int off = 128; off > 0; off >>= 1) {
      if (tid < off && rv[tid + off] < rv[tid]) { rv[tid] = rv[tid + off]; ri[tid] = ri[tid + off]; }
      __syncthreads();
    }
    if (tid == 0) { selI[k] = ri[0]; eL[ri[0]] = 3.0e38f; }
    __syncthreads();
  }

  if (tid < NH * NA) {
    int t = tid / NA, a = tid % NA;
    float vals[TOPK], sum = 0.0f;
    for (int k = 0; k < TOPK; ++k) {
      vals[k] = samples[(((size_t)b * NS + (size_t)selI[k]) * NH + t) * NA + a];
      sum += vals[k];
    }
    float mn = sum * (1.0f / (float)TOPK);
    float var = 0.0f;
    for (int k = 0; k < TOPK; ++k) { float d = vals[k] - mn; var += d * d; }
    float sd = __builtin_amdgcn_sqrtf(var * (1.0f / (float)(TOPK - 1)));
    int idx = (b * NH + t) * NA + a;
    actMean[idx] = 0.85f * mn + 0.15f * actMean[idx];
    actStd[idx]  = 0.85f * sd + 0.15f * actStd[idx];
  }
}

__global__ __launch_bounds__(256) void writeout_kernel(const float* __restrict__ actMean,
                                                       float* __restrict__ out) {
  int i = threadIdx.x;
  if (i < NB * NA) {                       // action_mean[:,0,:]
    int b = i / NA, a = i % NA;
    out[i] = actMean[b * NH * NA + a];
  } else if (i < NB * NA + NB * NH * NA) { // full action_mean
    out[i] = actMean[i - NB * NA];
  }
}

// ---------------------------------------------------------------------------
extern "C" void kernel_launch(void* const* d_in, const int* in_sizes, int n_in,
                              void* d_out, int out_size, void* d_ws, size_t ws_size,
                              hipStream_t stream) {
  const float* z_current = (const float*)d_in[0];  // [8,512]
  const float* z_goal    = (const float*)d_in[1];  // [8,512]
  const float* Wz        = (const float*)d_in[2];  // [512,512]
  const float* Wa        = (const float*)d_in[3];  // [7,512]
  const float* bias      = (const float*)d_in[4];  // [512]

  char* ws = (char*)d_ws;
  size_t off = 0;
  unsigned short* WzP = (unsigned short*)(ws + off); off += (size_t)KT1 * 32 * 32 * 16 * 2; // 544KB
  unsigned short* B0P = (unsigned short*)(ws + off); off += (size_t)NB * 32 * 32 * 16 * 2;  // 256KB
  float* zWz0     = (float*)(ws + off); off += (size_t)NB * ND * 4;                         // 16KB
  float* actMean  = (float*)(ws + off); off += 512;
  float* actStd   = (float*)(ws + off); off += 512;
  float* energies = (float*)(ws + off); off += (size_t)NB * NS * 4;                         // 128KB
  float* samples  = (float*)(ws + off); off += (size_t)NB * NS * NH * NA * 4;               // ~1.75MB

  prep_wz_kernel<<<(KT1 * 32 * 32 * 16) / 256, 256, 0, stream>>>(Wz, Wa, bias, WzP);
  init_kernel<<<NB, ND, 0, stream>>>(z_current, Wz, bias, zWz0, actMean, actStd);
  pack_b0_kernel<<<(NB * 32 * 32 * 16) / 256, 256, 0, stream>>>(Wa, zWz0, B0P);

  const int roll_blocks = NB * (NS / S_PER_BLK);   // 1024
  for (int it = 0; it < NITERS; ++it) {
    rollout_kernel<<<roll_blocks, ROLL_THREADS, 0, stream>>>(
        z_goal, WzP, B0P, actMean, actStd, samples, energies, it);
    topk_update_kernel<<<NB, 256, 0, stream>>>(energies, samples, actMean, actStd);
  }

  writeout_kernel<<<1, 256, 0, stream>>>(actMean, (float*)d_out);
}
